// SelfInteraction_81466939670710
// MI455X (gfx1250) — matrix operand order repaired and checked by reading, not compiled, not run on hardware
//
#include <hip/hip_runtime.h>
#include <math.h>

typedef __bf16 bf16_t;
typedef __attribute__((ext_vector_type(16))) __bf16 v16bf;
typedef __attribute__((ext_vector_type(8)))  __bf16 v8bf;
typedef __attribute__((ext_vector_type(8)))  float  v8f;
typedef __attribute__((ext_vector_type(4)))  float  v4f;

#define INV_SQRT3 0.57735026918962576451f

// ---------------- per-wave LDS layout (bytes) ----------------
#define OFF_S     0        // 16x128 f32  (s, kept fp32 across layers)
#define OFF_V     8192     // 16x384 f32  (v interleaved [ch][c], fp32)
#define OFF_SCHI  32768    // 16x392 bf16 (scal hi)   | reused: vecA hi (16x264)
#define OFF_SCLO  45312    // 16x392 bf16 (scal lo)   | reused: vecA lo (16x264)
#define OFF_H     57856    // 16x392 bf16 (h)         | reused: ys 16x128 f32
#define OFF_G     70400    // 16x256 bf16 (vec gates)
#define WAVE_LDS  78592
#define SCAL_STR  392
#define VA_STR    264

#define W1_ELEMS  147456   // 384x384
#define W2_ELEMS  245760   // 384x640  (quad gate columns 640..767 are dead code)
#define WS_ELEMS  49152    // 384x128
#define WV_ELEMS  32768    // 256x128
#define LAYER_ELEMS (2u*(W1_ELEMS+W2_ELEMS+WS_ELEMS+WV_ELEMS))   // hi+lo = 950272

__device__ __forceinline__ v8f wmma_bf(v16bf a, v16bf b, v8f c) {
  return __builtin_amdgcn_wmma_f32_16x16x32_bf16(false, a, false, b, (short)0, c,
                                                 false, false);
}

__device__ __forceinline__ void split_bf(float f, bf16_t& hi, bf16_t& lo) {
  hi = (bf16_t)f;
  lo = (bf16_t)(f - (float)hi);
}

__device__ __forceinline__ float silu(float x) {
  return x / (1.0f + __expf(-x));
}

// A fragment (16x32 bf16) from LDS, row-major with given stride.
// Layout (ISA 7.12.2): lanes 0-15: M=lane, K = kc*32 + {0..7, 16..23};
//                      lanes 16-31: M=lane-16, K = kc*32 + {8..15, 24..31}.
__device__ __forceinline__ v16bf load_afrag(const bf16_t* base, int stride,
                                            int lane, int kc) {
  int m  = lane & 15;
  int k0 = kc * 32 + ((lane >> 4) & 1) * 8;
  const bf16_t* p = base + m * stride + k0;
  v8bf a0 = *(const v8bf*)(p);
  v8bf a1 = *(const v8bf*)(p + 16);
  return __builtin_shufflevector(a0, a1, 0,1,2,3,4,5,6,7,8,9,10,11,12,13,14,15);
}

// B fragment (32x16 bf16) from pre-packed global weights: one 32B load/lane.
__device__ __forceinline__ v16bf load_bfrag(const bf16_t* __restrict__ packed,
                                            int KC, int jt, int kc, int lane) {
  const bf16_t* p = packed + ((size_t)((jt * KC + kc) * 32 + lane) << 4);
  return *(const v16bf*)p;
}

// -------- weight prep: fold scale, split hi/lo, pack into B-fragment order ----
__global__ void pack_weights(const float* __restrict__ src, bf16_t* __restrict__ hi,
                             bf16_t* __restrict__ lo, int K, int N, int Nsrc,
                             float scale) {
  int t = blockIdx.x * 256 + threadIdx.x;
  if (t >= K * N) return;
  int k = t / N, n = t - k * N;
  float f = src[(size_t)k * Nsrc + n] * scale;
  bf16_t h, l;
  split_bf(f, h, l);
  int jt = n >> 4, kc = k >> 5, kk = k & 31;
  int lane = (n & 15) | (((kk >> 3) & 1) << 4);
  int idx  = (kk & 7) | ((kk >> 4) << 3);
  int KC   = K >> 5;
  size_t off = ((size_t)((jt * KC + kc) * 32 + lane) << 4) + idx;
  hi[off] = h;
  lo[off] = l;
}

// ---------------- fused 2-block equivariant MLP ----------------
__global__ void __launch_bounds__(128)
fused_selfinteraction(const float* __restrict__ x, float* __restrict__ out,
                      const bf16_t* __restrict__ wpack) {
  extern __shared__ char smem_raw[];
  const int wave = threadIdx.x >> 5;
  const int lane = threadIdx.x & 31;
  char* wbase = smem_raw + wave * WAVE_LDS;

  float*  Ls    = (float*)(wbase + OFF_S);
  float*  Lv    = (float*)(wbase + OFF_V);
  bf16_t* Lsch  = (bf16_t*)(wbase + OFF_SCHI);
  bf16_t* Lscl  = (bf16_t*)(wbase + OFF_SCLO);
  bf16_t* Lh    = (bf16_t*)(wbase + OFF_H);
  bf16_t* Lg    = (bf16_t*)(wbase + OFF_G);
  bf16_t* LvAh  = (bf16_t*)(wbase + OFF_SCHI);  // reuse after GEMM3
  bf16_t* LvAl  = (bf16_t*)(wbase + OFF_SCLO);
  float*  Lys   = (float*)(wbase + OFF_H);      // reuse after GEMM2

  const int r0 = (blockIdx.x * 4 + wave) * 16;
  const int m16  = lane & 15;
  const int hf   = lane >> 4;       // half-wave id
  const int ncol = lane & 15;       // C-tile column
  const int mrow = hf * 8;          // C-tile row base

  // ---- stage 0: load 16 rows of x into fp32 LDS (s | v) ----
  {
    const float* xr = x + (size_t)r0 * 512;
    #pragma unroll 4
    for (int i = 0; i < 64; ++i) {
      int e4 = i * 32 + lane;
      int r  = e4 >> 7;             // 128 float4 per 512-f32 row
      int c4 = (e4 & 127) * 4;
      v4f d = *(const v4f*)(xr + (size_t)r * 512 + c4);
      if (c4 < 128) *(v4f*)(Ls + r * 128 + c4)         = d;
      else          *(v4f*)(Lv + r * 384 + (c4 - 128)) = d;
    }
  }

  for (int layer = 0; layer < 2; ++layer) {
    const bf16_t* Wb  = wpack + (size_t)layer * LAYER_ELEMS;
    const bf16_t* W1h = Wb;               const bf16_t* W1l = W1h + W1_ELEMS;
    const bf16_t* W2h = W1l + W1_ELEMS;   const bf16_t* W2l = W2h + W2_ELEMS;
    const bf16_t* Wsh = W2l + W2_ELEMS;   const bf16_t* Wsl = Wsh + WS_ELEMS;
    const bf16_t* Wvh = Wsl + WS_ELEMS;   const bf16_t* Wvl = Wvh + WV_ELEMS;

    // ---- build scal = [s | s^2 | vv] as bf16 hi/lo ----
    for (int i = 0; i < 64; ++i) {
      int k = hf * 64 + i;
      float sv = Ls[m16 * 128 + k];
      bf16_t h, l;
      split_bf(sv, h, l);
      Lsch[m16 * SCAL_STR + k] = h;  Lscl[m16 * SCAL_STR + k] = l;
      split_bf(sv * sv, h, l);
      Lsch[m16 * SCAL_STR + 128 + k] = h;  Lscl[m16 * SCAL_STR + 128 + k] = l;
      float vx = Lv[m16 * 384 + k * 3 + 0];
      float vy = Lv[m16 * 384 + k * 3 + 1];
      float vz = Lv[m16 * 384 + k * 3 + 2];
      split_bf((vx * vx + vy * vy + vz * vz) * INV_SQRT3, h, l);
      Lsch[m16 * SCAL_STR + 256 + k] = h;  Lscl[m16 * SCAL_STR + 256 + k] = l;
    }

    // ---- GEMM1: h = silu(scal @ W1p), 24 N-tiles, K=384 (3-term split) ----
    for (int jt = 0; jt < 24; ++jt) {
      v8f acc = {0,0,0,0,0,0,0,0};
      #pragma unroll 3
      for (int kc = 0; kc < 12; ++kc) {
        v16bf ah = load_afrag(Lsch, SCAL_STR, lane, kc);
        v16bf al = load_afrag(Lscl, SCAL_STR, lane, kc);
        v16bf bh = load_bfrag(W1h, 12, jt, kc, lane);
        v16bf bl = load_bfrag(W1l, 12, jt, kc, lane);
        acc = wmma_bf(ah, bh, acc);
        acc = wmma_bf(ah, bl, acc);
        acc = wmma_bf(al, bh, acc);
      }
      #pragma unroll
      for (int g = 0; g < 8; ++g)
        Lh[(g + mrow) * SCAL_STR + jt * 16 + ncol] = (bf16_t)silu(acc[g]);
    }

    // ---- GEMM2: g = silu(h @ W2p) (640 live cols); gate scal / stash vec gates
    for (int jt = 0; jt < 40; ++jt) {
      v8f acc = {0,0,0,0,0,0,0,0};
      #pragma unroll 3
      for (int kc = 0; kc < 12; ++kc) {
        v16bf ah = load_afrag(Lh, SCAL_STR, lane, kc);
        v16bf bh = load_bfrag(W2h, 12, jt, kc, lane);
        v16bf bl = load_bfrag(W2l, 12, jt, kc, lane);
        acc = wmma_bf(ah, bh, acc);
        acc = wmma_bf(ah, bl, acc);
      }
      if (jt < 24) {
        #pragma unroll
        for (int g = 0; g < 8; ++g) {
          int m = g + mrow, col = jt * 16 + ncol;
          float f = (float)Lsch[m * SCAL_STR + col] + (float)Lscl[m * SCAL_STR + col];
          f *= silu(acc[g]);
          bf16_t h, l;
          split_bf(f, h, l);
          Lsch[m * SCAL_STR + col] = h;
          Lscl[m * SCAL_STR + col] = l;
        }
      } else {
        #pragma unroll
        for (int g = 0; g < 8; ++g)
          Lg[(g + mrow) * 256 + (jt - 24) * 16 + ncol] = (bf16_t)silu(acc[g]);
      }
    }

    // ---- GEMM3: ys = scal_gated @ Wsp -> park in Lys (s must stay original) --
    for (int jt = 0; jt < 8; ++jt) {
      v8f acc = {0,0,0,0,0,0,0,0};
      #pragma unroll 3
      for (int kc = 0; kc < 12; ++kc) {
        v16bf ah = load_afrag(Lsch, SCAL_STR, lane, kc);
        v16bf al = load_afrag(Lscl, SCAL_STR, lane, kc);
        v16bf bh = load_bfrag(Wsh, 12, jt, kc, lane);
        v16bf bl = load_bfrag(Wsl, 12, jt, kc, lane);
        acc = wmma_bf(ah, bh, acc);
        acc = wmma_bf(ah, bl, acc);
        acc = wmma_bf(al, bh, acc);
      }
      #pragma unroll
      for (int g = 0; g < 8; ++g)
        Lys[(g + mrow) * 128 + jt * 16 + ncol] = acc[g];
    }

    // ---- GEMM4: yv_c = (gated vec)_c @ Wvp per component; v += yv in-place --
    for (int c = 0; c < 3; ++c) {
      for (int i = 0; i < 128; ++i) {         // build gated A_c (hi/lo)
        int ch = hf * 128 + i;
        float base;
        if (ch < 128) {
          base = Lv[m16 * 384 + ch * 3 + c];
        } else {
          int q = ch - 128;
          base = Ls[m16 * 128 + q] * Lv[m16 * 384 + q * 3 + c];
        }
        float a = base * (float)Lg[m16 * 256 + ch];
        bf16_t h, l;
        split_bf(a, h, l);
        LvAh[m16 * VA_STR + ch] = h;
        LvAl[m16 * VA_STR + ch] = l;
      }
      for (int jt = 0; jt < 8; ++jt) {
        v8f acc = {0,0,0,0,0,0,0,0};
        #pragma unroll
        for (int kc = 0; kc < 8; ++kc) {
          v16bf ah = load_afrag(LvAh, VA_STR, lane, kc);
          v16bf al = load_afrag(LvAl, VA_STR, lane, kc);
          v16bf bh = load_bfrag(Wvh, 8, jt, kc, lane);
          v16bf bl = load_bfrag(Wvl, 8, jt, kc, lane);
          acc = wmma_bf(ah, bh, acc);
          acc = wmma_bf(ah, bl, acc);
          acc = wmma_bf(al, bh, acc);
        }
        #pragma unroll
        for (int g = 0; g < 8; ++g) {
          int m = g + mrow, col = jt * 16 + ncol;
          Lv[m * 384 + col * 3 + c] += acc[g];
        }
      }
    }

    // ---- s residual ----
    for (int i = 0; i < 64; ++i) {
      int k = hf * 64 + i;
      Ls[m16 * 128 + k] += Lys[m16 * 128 + k];
    }

    // ---- equivariant layernorm between blocks ----
    if (layer == 0) {
      float sum = 0.f, sum2 = 0.f;
      for (int i = 0; i < 64; ++i) {
        float t = Ls[m16 * 128 + hf * 64 + i];
        sum += t;  sum2 += t * t;
      }
      sum  += __shfl_xor(sum, 16);
      sum2 += __shfl_xor(sum2, 16);
      float mu  = sum * (1.0f / 128.0f);
      float var = sum2 * (1.0f / 128.0f) - mu * mu;
      float inv = 1.0f / sqrtf(var + 1e-6f);
      float vs = 0.f;
      for (int i = 0; i < 192; ++i) {
        float t = Lv[m16 * 384 + hf * 192 + i];
        vs += t * t;
      }
      vs += __shfl_xor(vs, 16);
      float vinv = 1.0f / sqrtf(vs * (1.0f / 384.0f) + 1e-6f);
      for (int i = 0; i < 64; ++i) {
        int k = hf * 64 + i;
        Ls[m16 * 128 + k] = (Ls[m16 * 128 + k] - mu) * inv;
      }
      for (int i = 0; i < 192; ++i) {
        int f = hf * 192 + i;
        Lv[m16 * 384 + f] *= vinv;
      }
    }
  }

  // ---- write out [s | v] fp32 ----
  {
    float* orow = out + (size_t)r0 * 512;
    #pragma unroll 4
    for (int i = 0; i < 64; ++i) {
      int e4 = i * 32 + lane;
      int r  = e4 >> 7;
      int c4 = (e4 & 127) * 4;
      v4f d;
      if (c4 < 128) d = *(const v4f*)(Ls + r * 128 + c4);
      else          d = *(const v4f*)(Lv + r * 384 + (c4 - 128));
      *(v4f*)(orow + (size_t)r * 512 + c4) = d;
    }
  }
}

extern "C" void kernel_launch(void* const* d_in, const int* in_sizes, int n_in,
                              void* d_out, int out_size, void* d_ws, size_t ws_size,
                              hipStream_t stream) {
  const float* x = (const float*)d_in[0];
  // d_in[1] = mask (all-true, unused by reference math), d_in[2] = coord (unused)
  const float* W[2][4] = {
      {(const float*)d_in[3], (const float*)d_in[4], (const float*)d_in[5], (const float*)d_in[6]},
      {(const float*)d_in[7], (const float*)d_in[8], (const float*)d_in[9], (const float*)d_in[10]}};

  bf16_t* ws = (bf16_t*)d_ws;
  size_t o = 0;
  const float s384 = 1.0f / sqrtf(384.0f);
  const float s256 = 1.0f / sqrtf(256.0f);

  auto prep = [&](const float* src, int K, int N, int Nsrc, float scale) {
    bf16_t* hi = ws + o;  o += (size_t)K * N;
    bf16_t* lo = ws + o;  o += (size_t)K * N;
    int tot = K * N;
    pack_weights<<<(tot + 255) / 256, 256, 0, stream>>>(src, hi, lo, K, N, Nsrc, scale);
  };

  for (int l = 0; l < 2; ++l) {
    prep(W[l][0], 384, 384, 384, s384);   // W1
    prep(W[l][1], 384, 640, 768, s384);   // W2 (live 640 cols; quad gates dead)
    prep(W[l][2], 384, 128, 128, s384);   // Ws
    prep(W[l][3], 256, 128, 128, s256);   // Wv
  }

  const size_t lds_bytes = 4u * WAVE_LDS;  // 314368 B, 4 wave32 per block
  fused_selfinteraction<<<65536 / 64, 128, lds_bytes, stream>>>(
      x, (float*)d_out, ws);
}
